// SkeletonVQVAE_10342281249144
// MI455X (gfx1250) — compile-verified, matrix operand order checked by reading, not run on hardware
//
#include <hip/hip_runtime.h>

typedef __attribute__((ext_vector_type(16))) __bf16 v16bf;
typedef __attribute__((ext_vector_type(8)))  float  v8f;

#define AS 132  // activation row stride in floats (16 rows per wave tile)

// ---- LDS layout -------------------------------------------------------------
// u32 region: pre-swizzled bf16 weight fragments (each u32 = 2 bf16, B-operand order)
#define W0_U 0        // ew0  K=128(pad from 104) N=128 : 4x8 tiles -> 8192 u32
#define W1_U 8192     // ew1  K=128 N=64  : 4x4 -> 4096
#define W2_U 12288    // ew2  K=64  N=32  : 2x2 -> 1024
#define WC_U 13312    // codebook^T K=32 N=64 : 1x4 -> 1024
#define W3_U 14336    // dw0  K=32  N=64  : 1x4 -> 1024
#define W4_U 15360    // dw1  K=64  N=128 : 2x8 -> 4096
#define W5_U 19456    // dw2  K=128 N=112(pad from 104) : 4x7 -> 7168
#define WTOT_U 26624
// float region (offsets in floats, region starts at byte WTOT_U*4)
#define F_EB0 0
#define F_EG0 128
#define F_EBT0 256
#define F_EB1 384
#define F_EG1 448
#define F_EBT1 512
#define F_EB2 576
#define F_EG2 608
#define F_EBT2 640
#define F_DB0 672
#define F_DG0 736
#define F_DBT0 800
#define F_DB1 864
#define F_DG1 992
#define F_DBT1 1120
#define F_DB2 1248   // 112 (padded from 104)
#define F_CN2 1360   // 64  ||codebook_n||^2
#define F_CBK 1424   // 2048 codebook f32 [64][32]
#define F_ACT 3472   // 4 waves * 16*AS = 8448
#define F_IDX 11920  // 4 waves * 16 int
#define F_SSE 11984  // 4 partial sums
#define F_TOT 11988
#define SMEM_BYTES (WTOT_U * 4 + F_TOT * 4)  // 154448 bytes

union Frag {
    v16bf    v;
    uint4    u4[2];
    unsigned u[8];
};

// cold-path pack (staging only)
static __device__ __forceinline__ unsigned short f2bf(float f) {
    unsigned u;
    __builtin_memcpy(&u, &f, 4);
    unsigned r = 0x7FFFu + ((u >> 16) & 1u);  // round-to-nearest-even
    return (unsigned short)((u + r) >> 16);
}
static __device__ __forceinline__ unsigned pk2(float a, float b) {
    return (unsigned)f2bf(a) | ((unsigned)f2bf(b) << 16);
}

// Stage a weight matrix (global f32, logical KxN; trans => source is [N][K])
// into LDS in B-fragment-linear order: index = ((kt*Nt+nt)*32+lane)*8+v,
// element (k,n): k = kt*32 + (v>>2)*16 + (lane>>4)*8 + (v&3)*2 (+0/+1), n = nt*16 + (lane&15)
static __device__ void stage_weight(const float* __restrict__ g, unsigned* dst,
                                    int Ksrc, int Nsrc, int Kt, int Nt, bool trans,
                                    int tid, int nthr) {
    const int total = Kt * Nt * 256;
    for (int f = tid; f < total; f += nthr) {
        int v = f & 7, lane = (f >> 3) & 31, t = f >> 8;
        int nt = t % Nt, kt = t / Nt;
        int n = nt * 16 + (lane & 15);
        int k = kt * 32 + ((v >> 2) << 4) + ((lane >> 4) << 3) + ((v & 3) << 1);
        float a = 0.f, b = 0.f;
        if (n < Nsrc) {
            if (!trans) {
                if (k < Ksrc) a = g[(size_t)k * Nsrc + n];
                if (k + 1 < Ksrc) b = g[(size_t)(k + 1) * Nsrc + n];
            } else {
                if (k < Ksrc) a = g[(size_t)n * Ksrc + k];
                if (k + 1 < Ksrc) b = g[(size_t)n * Ksrc + k + 1];
            }
        }
        dst[f] = pk2(a, b);
    }
}

static __device__ __forceinline__ void copyv(const float* __restrict__ g, float* d, int n,
                                             int tid, int nthr) {
    for (int i = tid; i < n; i += nthr) d[i] = g[i];
}

// One WMMA-tiled matmul layer: actf cols [0,KT*32) f32 in -> cols [colOfs, colOfs+NT*16) f32 out.
// A fragments fully loaded to registers before any C store, so in-place is safe.
// B fragments are fragment-linear in LDS and double-buffered in registers: loads for tile nt+1
// are issued before the WMMAs of tile nt, so LDS latency is covered by KT WMMAs.
template <int KT, int NT>
static __device__ __forceinline__ void layer_mm(float* actf, const unsigned* wf, int lane,
                                                int colOfs) {
    const int row = lane & 15, half = lane >> 4;
    const float* abase = actf + row * AS + (half << 3);
    Frag a[KT];
#pragma unroll
    for (int kt = 0; kt < KT; ++kt) {
#pragma unroll
        for (int v = 0; v < 8; ++v) {
            const int k0 = kt * 32 + ((v >> 2) << 4) + ((v & 3) << 1);
            float2 f = *(const float2*)(abase + k0);
            a[kt].v[2 * v]     = (__bf16)f.x;   // native v_cvt_pk_bf16_f32 path
            a[kt].v[2 * v + 1] = (__bf16)f.y;
        }
    }
    const unsigned* wbase = wf + (lane << 3);
    Frag b[2][KT];
#pragma unroll
    for (int kt = 0; kt < KT; ++kt) {
        const uint4* wp = (const uint4*)(wbase + ((kt * NT) << 8));
        b[0][kt].u4[0] = wp[0];
        b[0][kt].u4[1] = wp[1];
    }
#pragma unroll
    for (int nt = 0; nt < NT; ++nt) {
        if (nt + 1 < NT) {
#pragma unroll
            for (int kt = 0; kt < KT; ++kt) {
                const uint4* wp = (const uint4*)(wbase + ((kt * NT + nt + 1) << 8));
                b[(nt + 1) & 1][kt].u4[0] = wp[0];
                b[(nt + 1) & 1][kt].u4[1] = wp[1];
            }
        }
        v8f acc;
#pragma unroll
        for (int i = 0; i < 8; ++i) acc[i] = 0.f;
#pragma unroll
        for (int kt = 0; kt < KT; ++kt) {
            acc = __builtin_amdgcn_wmma_f32_16x16x32_bf16(false, a[kt].v, false, b[nt & 1][kt].v,
                                                          (short)0, acc, false, false);
        }
        float* cbase = actf + (half << 3) * AS + colOfs + nt * 16 + row;
#pragma unroll
        for (int v = 0; v < 8; ++v) cbase[v * AS] = acc[v];
    }
}

// bias add + LayerNorm + optional activation over cols [0,N); 2 lanes per row.
// ACT: 0=none, 1=ReLU, 2=exact GELU
template <int N, int ACT>
static __device__ __forceinline__ void post_ln_act(float* actf, const float* bias,
                                                   const float* gain, const float* beta,
                                                   int lane) {
    const int row = lane & 15, half = lane >> 4;
    const int c0 = half * (N / 2);
    float* rp = actf + row * AS;
    float s = 0.f, s2 = 0.f;
#pragma unroll 4
    for (int c = c0; c < c0 + N / 2; ++c) {
        float x = rp[c] + bias[c];
        rp[c] = x;
        s += x;
        s2 += x * x;
    }
    s += __shfl_xor(s, 16);
    s2 += __shfl_xor(s2, 16);
    const float inv = 1.0f / (float)N;
    float m = s * inv;
    float var = s2 * inv - m * m;
    float rstd = rsqrtf(var + 1e-5f);
#pragma unroll 4
    for (int c = c0; c < c0 + N / 2; ++c) {
        float x = (rp[c] - m) * rstd * gain[c] + beta[c];
        if (ACT == 1) x = fmaxf(x, 0.f);
        if (ACT == 2) x = 0.5f * x * (1.f + erff(x * 0.70710678118654752f));
        rp[c] = x;
    }
}

struct VArgs {
    const float *x;
    const float *ew0, *eb0, *eg0, *ebt0;
    const float *ew1, *eb1, *eg1, *ebt1;
    const float *ew2, *eb2, *eg2, *ebt2;
    const float *cbk;
    const float *dw0, *db0, *dg0, *dbt0;
    const float *dw1, *db1, *dg1, *dbt1;
    const float *dw2, *db2;
    float* out;
    int* idxOut;
    float* sseAcc;
};

__global__ __launch_bounds__(128) void vqvae_main(VArgs A, int itersPerWave) {
    extern __shared__ char smem[];
    unsigned* wb = (unsigned*)smem;
    float* fb = (float*)(smem + WTOT_U * 4);
    const int tid = threadIdx.x, lane = tid & 31, wave = tid >> 5;

    // ---- stage weights (pre-swizzled bf16 B fragments) + params into LDS ----
    stage_weight(A.ew0, wb + W0_U, 104, 128, 4, 8, false, tid, 128);
    stage_weight(A.ew1, wb + W1_U, 128, 64, 4, 4, false, tid, 128);
    stage_weight(A.ew2, wb + W2_U, 64, 32, 2, 2, false, tid, 128);
    stage_weight(A.cbk, wb + WC_U, 32, 64, 1, 4, true, tid, 128);  // codebook^T
    stage_weight(A.dw0, wb + W3_U, 32, 64, 1, 4, false, tid, 128);
    stage_weight(A.dw1, wb + W4_U, 64, 128, 2, 8, false, tid, 128);
    stage_weight(A.dw2, wb + W5_U, 128, 104, 4, 7, false, tid, 128);
    copyv(A.eb0, fb + F_EB0, 128, tid, 128);
    copyv(A.eg0, fb + F_EG0, 128, tid, 128);
    copyv(A.ebt0, fb + F_EBT0, 128, tid, 128);
    copyv(A.eb1, fb + F_EB1, 64, tid, 128);
    copyv(A.eg1, fb + F_EG1, 64, tid, 128);
    copyv(A.ebt1, fb + F_EBT1, 64, tid, 128);
    copyv(A.eb2, fb + F_EB2, 32, tid, 128);
    copyv(A.eg2, fb + F_EG2, 32, tid, 128);
    copyv(A.ebt2, fb + F_EBT2, 32, tid, 128);
    copyv(A.db0, fb + F_DB0, 64, tid, 128);
    copyv(A.dg0, fb + F_DG0, 64, tid, 128);
    copyv(A.dbt0, fb + F_DBT0, 64, tid, 128);
    copyv(A.db1, fb + F_DB1, 128, tid, 128);
    copyv(A.dg1, fb + F_DG1, 128, tid, 128);
    copyv(A.dbt1, fb + F_DBT1, 128, tid, 128);
    for (int i = tid; i < 112; i += 128) fb[F_DB2 + i] = (i < 104) ? A.db2[i] : 0.f;
    copyv(A.cbk, fb + F_CBK, 64 * 32, tid, 128);
    if (tid < 64) {
        float s = 0.f;
        for (int k = 0; k < 32; ++k) {
            float v = A.cbk[tid * 32 + k];
            s += v * v;
        }
        fb[F_CN2 + tid] = s;
    }
    __syncthreads();

    float* actf = fb + F_ACT + wave * (16 * AS);
    int* idxw = (int*)(fb + F_IDX) + wave * 16;
    const float* cn2 = fb + F_CN2;
    const float* cbk = fb + F_CBK;
    float sse = 0.f;

    for (int it = 0; it < itersPerWave; ++it) {
        const int tileIdx = blockIdx.x * (4 * itersPerWave) + it * 4 + wave;
        const size_t rowBase = (size_t)tileIdx * 16;

        // ---- load x tile [16 x 104] (contiguous float4) + zero K-pad cols 104..127
        {
            const float4* ip = (const float4*)(A.x + rowBase * 104);
            for (int t = lane; t < 416; t += 32) {
                float4 f = ip[t];
                int b = t * 4;
                actf[((b + 0) / 104) * AS + (b + 0) % 104] = f.x;
                actf[((b + 1) / 104) * AS + (b + 1) % 104] = f.y;
                actf[((b + 2) / 104) * AS + (b + 2) % 104] = f.z;
                actf[((b + 3) / 104) * AS + (b + 3) % 104] = f.w;
            }
            for (int t = lane; t < 16 * 24; t += 32) actf[(t / 24) * AS + 104 + (t % 24)] = 0.f;
        }
        // ---- prefetch next x tile for this wave (+4 tiles = +64 rows = 26624 B)
        if (it + 1 < itersPerWave) {
            const char* nx = (const char*)(A.x + (rowBase + 64) * 104);
            __builtin_prefetch(nx + lane * 128, 0, 3);
            __builtin_prefetch(nx + 4096 + lane * 128, 0, 3);
        }
        __syncthreads();

        // ---- encoder
        layer_mm<4, 8>(actf, wb + W0_U, lane, 0);
        __syncthreads();
        post_ln_act<128, 1>(actf, fb + F_EB0, fb + F_EG0, fb + F_EBT0, lane);
        __syncthreads();
        layer_mm<4, 4>(actf, wb + W1_U, lane, 0);
        __syncthreads();
        post_ln_act<64, 1>(actf, fb + F_EB1, fb + F_EG1, fb + F_EBT1, lane);
        __syncthreads();
        layer_mm<2, 2>(actf, wb + W2_U, lane, 0);
        __syncthreads();
        post_ln_act<32, 0>(actf, fb + F_EB2, fb + F_EG2, fb + F_EBT2, lane);
        __syncthreads();

        // ---- VQ: zc = z @ codebook^T into cols 64..127 (z kept in cols 0..31)
        layer_mm<1, 4>(actf, wb + WC_U, lane, 64);
        __syncthreads();
        {
            int row = lane & 15;
            float best = 3.4e38f;
            int bi = 0;
            const float4* zp = (const float4*)(actf + row * AS + 64);
            const float4* cp = (const float4*)cn2;
#pragma unroll 4
            for (int q = 0; q < 16; ++q) {
                float4 zc = zp[q];
                float4 c2 = cp[q];
                float d0 = c2.x - 2.0f * zc.x;
                float d1 = c2.y - 2.0f * zc.y;
                float d2 = c2.z - 2.0f * zc.z;
                float d3 = c2.w - 2.0f * zc.w;
                if (d0 < best) { best = d0; bi = 4 * q + 0; }
                if (d1 < best) { best = d1; bi = 4 * q + 1; }
                if (d2 < best) { best = d2; bi = 4 * q + 2; }
                if (d3 < best) { best = d3; bi = 4 * q + 3; }
            }
            if (lane < 16) {
                idxw[row] = bi;
                A.idxOut[rowBase + row] = bi;
            }
        }
        __syncthreads();
        // commit SSE + straight-through: quant = zq  (float4 over [16 x 32])
        for (int t = lane; t < 128; t += 32) {
            int r = t >> 3, c = (t & 7) << 2;
            float4 z4 = *(const float4*)(actf + r * AS + c);
            float4 q4 = *(const float4*)(cbk + idxw[r] * 32 + c);
            float dx = q4.x - z4.x, dy = q4.y - z4.y, dz = q4.z - z4.z, dw = q4.w - z4.w;
            sse += dx * dx + dy * dy + dz * dz + dw * dw;
            *(float4*)(actf + r * AS + c) = q4;
        }
        __syncthreads();

        // ---- decoder
        layer_mm<1, 4>(actf, wb + W3_U, lane, 0);
        __syncthreads();
        post_ln_act<64, 2>(actf, fb + F_DB0, fb + F_DG0, fb + F_DBT0, lane);
        __syncthreads();
        layer_mm<2, 8>(actf, wb + W4_U, lane, 0);
        __syncthreads();
        post_ln_act<128, 2>(actf, fb + F_DB1, fb + F_DG1, fb + F_DBT1, lane);
        __syncthreads();
        layer_mm<4, 7>(actf, wb + W5_U, lane, 0);
        __syncthreads();

        // ---- write out [16 x 104] + bias (contiguous float4)
        {
            float4* op = (float4*)(A.out + rowBase * 104);
            for (int t = lane; t < 416; t += 32) {
                int b = t * 4;
                float4 f;
                f.x = actf[((b + 0) / 104) * AS + (b + 0) % 104] + fb[F_DB2 + (b + 0) % 104];
                f.y = actf[((b + 1) / 104) * AS + (b + 1) % 104] + fb[F_DB2 + (b + 1) % 104];
                f.z = actf[((b + 2) / 104) * AS + (b + 2) % 104] + fb[F_DB2 + (b + 2) % 104];
                f.w = actf[((b + 3) / 104) * AS + (b + 3) % 104] + fb[F_DB2 + (b + 3) % 104];
                op[t] = f;
            }
        }
        __syncthreads();
    }

    // ---- block-level commit-SSE reduction, one atomic per block
#pragma unroll
    for (int m = 16; m >= 1; m >>= 1) sse += __shfl_xor(sse, m);
    if (lane == 0) fb[F_SSE + wave] = sse;
    __syncthreads();
    if (tid == 0) {
        float s = fb[F_SSE] + fb[F_SSE + 1] + fb[F_SSE + 2] + fb[F_SSE + 3];
        atomicAdd(A.sseAcc, s);
    }
}

__global__ void vqvae_init(float* acc) {
    if (threadIdx.x == 0) acc[0] = 0.f;
}

__global__ __launch_bounds__(128) void vqvae_finalize(const float* __restrict__ cbk,
                                                      const float* __restrict__ sseAcc,
                                                      float* lossOut, float invCount) {
    __shared__ float invn[64];
    __shared__ float partial[128];
    int tid = threadIdx.x;
    if (tid < 64) {
        float s = 0.f;
        for (int k = 0; k < 32; ++k) {
            float v = cbk[tid * 32 + k];
            s += v * v;
        }
        invn[tid] = rsqrtf(s);
    }
    __syncthreads();
    float acc = 0.f;
    for (int p = tid; p < 4096; p += 128) {
        int i = p >> 6, j = p & 63;
        float d = 0.f;
        for (int k = 0; k < 32; ++k) d += cbk[i * 32 + k] * cbk[j * 32 + k];
        float cs = d * invn[i] * invn[j];
        acc += cs * cs;
    }
    partial[tid] = acc;
    __syncthreads();
    if (tid == 0) {
        float t = 0.f;
        for (int i = 0; i < 128; ++i) t += partial[i];
        float ortho = t / 4096.f - 1.f / 64.f;
        float commit = sseAcc[0] * invCount;
        lossOut[0] = 0.1f * commit + 1.0f * ortho;
    }
}

extern "C" void kernel_launch(void* const* d_in, const int* in_sizes, int n_in, void* d_out,
                              int out_size, void* d_ws, size_t ws_size, hipStream_t stream) {
    VArgs A;
    A.x = (const float*)d_in[0];
    A.ew0 = (const float*)d_in[1];
    A.eb0 = (const float*)d_in[2];
    A.eg0 = (const float*)d_in[3];
    A.ebt0 = (const float*)d_in[4];
    A.ew1 = (const float*)d_in[5];
    A.eb1 = (const float*)d_in[6];
    A.eg1 = (const float*)d_in[7];
    A.ebt1 = (const float*)d_in[8];
    A.ew2 = (const float*)d_in[9];
    A.eb2 = (const float*)d_in[10];
    A.eg2 = (const float*)d_in[11];
    A.ebt2 = (const float*)d_in[12];
    A.cbk = (const float*)d_in[13];
    A.dw0 = (const float*)d_in[14];
    A.db0 = (const float*)d_in[15];
    A.dg0 = (const float*)d_in[16];
    A.dbt0 = (const float*)d_in[17];
    A.dw1 = (const float*)d_in[18];
    A.db1 = (const float*)d_in[19];
    A.dg1 = (const float*)d_in[20];
    A.dbt1 = (const float*)d_in[21];
    A.dw2 = (const float*)d_in[22];
    A.db2 = (const float*)d_in[23];

    const int nRows = in_sizes[0] / 104;  // 524288
    A.out = (float*)d_out;
    A.idxOut = (int*)d_out + (size_t)nRows * 104 + 1;  // after out + vq_loss
    A.sseAcc = (float*)d_ws;
    float* lossOut = (float*)d_out + (size_t)nRows * 104;

    const int tiles = nRows / 16;  // 32768
    int iters = 16;
    int blocks = tiles / (4 * iters);
    if (blocks == 0 || blocks * 4 * iters != tiles) {
        iters = 1;
        blocks = tiles / 4;
    }

    vqvae_init<<<1, 32, 0, stream>>>(A.sseAcc);
    vqvae_main<<<blocks, 128, SMEM_BYTES, stream>>>(A, iters);
    vqvae_finalize<<<1, 128, 0, stream>>>(A.cbk, A.sseAcc, lossOut,
                                          1.0f / ((float)nRows * 32.0f));
}
